// ShiftReduceEncoder_28278064677622
// MI455X (gfx1250) — matrix operand order repaired and checked by reading, not compiled, not run on hardware
//
#include <hip/hip_runtime.h>
#include <hip/hip_bf16.h>
#include <math.h>

typedef __bf16 bf16_t;
typedef __attribute__((ext_vector_type(16))) bf16_t v16bf;
typedef __attribute__((ext_vector_type(8)))  float  v8f;

#define NSEQ   256
#define TSTEPS 511          // 2*N-1
#define NB     64           // persistent blocks
#define NT     (NB*256)     // 16384 threads
#define NWAVES (NT/32)      // 512 waves

// ---------------- workspace layout ----------------
// [ packed bf16 weights (hi/lo) | fp32 state | bf16 vectors | control ]
static constexpr size_t N_IH0 = (size_t)256*33*512;   // 4096 x 1056(pad of 1026)
static constexpr size_t N_HX  = (size_t)256*32*512;   // 4096 x 1024
static constexpr size_t N_RED = (size_t)64*192*512;   // 1024 x 6144
static constexpr size_t N_RES = (size_t)64*32*512;    // 1024 x 1024
static constexpr size_t O_IH0H=0,              O_IH0L=O_IH0H+N_IH0,
                        O_HH0H=O_IH0L+N_IH0,   O_HH0L=O_HH0H+N_HX,
                        O_IH1H=O_HH0L+N_HX,    O_IH1L=O_IH1H+N_HX,
                        O_HH1H=O_IH1L+N_HX,    O_HH1L=O_HH1H+N_HX,
                        O_REDH=O_HH1L+N_HX,    O_REDL=O_REDH+N_RED,
                        O_RESH=O_REDL+N_RED,   O_RESL=O_RESH+N_RES,
                        W_END =O_RESL+N_RES;
static constexpr size_t FB_BYTES = ((W_END*2 + 255)/256)*256;

static constexpr size_t F_HST=0,
                        F_CST =F_HST+(size_t)257*2048,
                        F_STK =F_CST+(size_t)257*2048,
                        F_G0  =F_STK+(size_t)NSEQ*1024,
                        F_G1  =F_G0+4096,
                        F_B0g =F_G1+4096,
                        F_B1g =F_B0g+4096,
                        F_RPRE=F_B1g+4096,
                        F_RESV=F_RPRE+1024,
                        F_LREC=F_RESV+1024,
                        F_AREC=F_LREC+2*TSTEPS,
                        F_END =F_AREC+TSTEPS;
static constexpr size_t VB_FLOATS = ((F_END+63)/64)*64;

static constexpr size_t V_X0H=0,     V_X0L=1056,
                        V_P0H=2112,  V_P0L=3136,
                        V_P1H=4160,  V_P1L=5184,
                        V_I1H=6208,  V_I1L=7232,
                        V_ZH =8256,  V_ZL =14400,
                        V_RH =20544, V_RL =21568,
                        V_END=22592;
static constexpr size_t CTL_BYTES_AFTER_VB = ((V_END*2+63)/64)*64;

__host__ __device__ static inline bf16_t*  ws_w(void* ws){ return (bf16_t*)ws; }
__host__ __device__ static inline float*   ws_f(void* ws){ return (float*)((char*)ws + FB_BYTES); }
__host__ __device__ static inline bf16_t*  ws_v(void* ws){ return (bf16_t*)(ws_f(ws)+VB_FLOATS); }
__host__ __device__ static inline unsigned* ws_c(void* ws){ return (unsigned*)((char*)ws_v(ws) + CTL_BYTES_AFTER_VB); }
// ctl slots: [0]=sp [1]=bi [2]=act [3]=row [4]=cnt [5]=gen

__device__ static inline float sigm(float x){ return 1.0f/(1.0f+expf(-x)); }
__device__ static inline void bsplit(float v, bf16_t* hi, bf16_t* lo){
  bf16_t h=(bf16_t)v; *hi=h; *lo=(bf16_t)(v-(float)h);
}

__device__ static inline v8f wmma_bf16(v16bf a, v16bf b, v8f c){
  return __builtin_amdgcn_wmma_f32_16x16x32_bf16(false, a, false, b, (short)0, c, false, false);
}

// One 16-row strip of y += W x over chunk range [c0,c1).
// Weights pre-packed per 16x32 tile: 32 lanes x 16 contiguous bf16 (A layout).
// B operand: x replicated across all 16 columns (lane<16 -> K=e, lane>=16 -> K=16+e).
__device__ static inline void gemv_tiles(const bf16_t* Whi, const bf16_t* Wlo,
                                         const bf16_t* xhi, const bf16_t* xlo,
                                         float* out, int strip, int c0, int c1,
                                         int chunks, int lane){
  v8f acc = {};
  int koff = (lane & 16) ? 16 : 0;
  for (int kc=c0; kc<c1; ++kc){
    size_t tb = ((size_t)strip*chunks + kc)*512 + (size_t)lane*16;
    v16bf ah = *(const v16bf*)(Whi + tb);
    v16bf al = *(const v16bf*)(Wlo + tb);
    v16bf bh = *(const v16bf*)(xhi + kc*32 + koff);
    v16bf bl = *(const v16bf*)(xlo + kc*32 + koff);
    acc = wmma_bf16(ah, bh, acc);   // hi*hi
    acc = wmma_bf16(ah, bl, acc);   // hi*lo
    acc = wmma_bf16(al, bh, acc);   // lo*hi
  }
  if ((lane & 15) == 0){            // lane0 -> M 0..7, lane16 -> M 8..15 (f32 C/D layout)
    int m0 = strip*16 + ((lane>>4)<<3);
#pragma unroll
    for (int v=0; v<8; ++v) atomicAdd(out + m0 + v, acc[v]);
  }
}

// device-wide sense barrier (persistent grid, all blocks resident)
__device__ static inline void gsync(unsigned* cnt, unsigned* gen){
  __syncthreads();
  __threadfence();                          // release my stores
  if (threadIdx.x == 0){
    unsigned old = atomicAdd(gen, 0u);
    unsigned a   = atomicAdd(cnt, 1u);
    if (a == (unsigned)(NB-1)){
      atomicExch(cnt, 0u);
      __threadfence();
      atomicAdd(gen, 1u);
    } else {
      while (atomicAdd(gen, 0u) == old) __builtin_amdgcn_s_sleep(1);
    }
  }
  __syncthreads();
  __threadfence();                          // acquire: invalidate stale lines
}

// ---------------- repack: fp32 row-major -> bf16 hi/lo WMMA A-tiles ----------------
__global__ void srenc_repack(const float* __restrict__ W, void* ws,
                             unsigned long long ohi, unsigned long long olo,
                             int cols, int chunks, int ntiles){
  int g = blockIdx.x*blockDim.x + threadIdx.x;
  int tile = g >> 5;
  if (tile >= ntiles) return;
  int lane = g & 31;
  int s = tile / chunks, kc = tile % chunks;
  int m = s*16 + (lane & 15);
  v16bf hv, lv;
#pragma unroll
  for (int e=0; e<16; ++e){
    int K = ((lane < 16) ? 0 : 16) + e;  // contiguous K order: lane<16 -> K=e, lane>=16 -> K=16+e
    K += kc*32;
    float w = (K < cols) ? W[(size_t)m*cols + K] : 0.0f;
    bf16_t h = (bf16_t)w;
    hv[e] = h;
    lv[e] = (bf16_t)(w - (float)h);
  }
  bf16_t* wb = ws_w(ws);
  size_t tb = (size_t)tile*512 + (size_t)lane*16;
  *(v16bf*)(wb + ohi + tb) = hv;
  *(v16bf*)(wb + olo + tb) = lv;
}

// ---------------- init: zero state, fold biases, reset control ----------------
__global__ void srenc_init(const float* bih0, const float* bhh0,
                           const float* bih1, const float* bhh1,
                           const float* h0, const float* c0, void* ws){
  float* fb = ws_f(ws);
  bf16_t* vb = ws_v(ws);
  unsigned* ctl = ws_c(ws);
  size_t stride = (size_t)gridDim.x*blockDim.x;
  size_t g = (size_t)blockIdx.x*blockDim.x + threadIdx.x;
  for (size_t i=g; i<(size_t)257*2048; i+=stride){
    fb[F_HST+i] = (i<2048) ? h0[i] : 0.0f;
    fb[F_CST+i] = (i<2048) ? c0[i] : 0.0f;
  }
  for (size_t i=g; i<(size_t)NSEQ*1024; i+=stride) fb[F_STK+i] = 0.0f;
  for (size_t i=g; i<4096; i+=stride){
    fb[F_B0g+i] = bih0[i] + bhh0[i];
    fb[F_B1g+i] = bih1[i] + bhh1[i];
    fb[F_G0+i] = 0.0f; fb[F_G1+i] = 0.0f;
  }
  for (size_t i=g; i<1024; i+=stride){ fb[F_RPRE+i]=0.0f; fb[F_RESV+i]=0.0f; }
  for (size_t i=g; i<(size_t)3*TSTEPS; i+=stride) fb[F_LREC+i] = 0.0f;  // LREC+AREC contiguous
  for (size_t i=g; i<V_END; i+=stride) vb[i] = (bf16_t)0.0f;
  if (g == 0){ ctl[0]=0; ctl[1]=0; ctl[2]=0; ctl[3]=0; ctl[4]=0; ctl[5]=0; }
}

// ---------------- main: persistent cooperative shift-reduce loop ----------------
__global__ __launch_bounds__(256) void srenc_main(const int* __restrict__ idx,
                                                  const float* __restrict__ embed,
                                                  const float* __restrict__ scw,
                                                  const float* __restrict__ scb,
                                                  const float* __restrict__ red_b1,
                                                  const float* __restrict__ res_b,
                                                  void* ws, float* __restrict__ out){
  bf16_t* wb = ws_w(ws);
  float*  fb = ws_f(ws);
  bf16_t* vb = ws_v(ws);
  unsigned* ctl = ws_c(ws);
  unsigned *cnt = ctl+4, *gen = ctl+5;
  const int tid = threadIdx.x, bid = blockIdx.x;
  const int gtid = bid*256 + tid;
  const int wave = gtid >> 5, lane = tid & 31;

  for (int t=0; t<TSTEPS; ++t){
    // ---- Phase A: scorer + action decision (single thread; 2x1024 fp32 MACs) ----
    if (bid==0 && tid==0){
      int sp = (int)ctl[0], bi = (int)ctl[1];
      const float* htop = fb + F_HST + (size_t)sp*2048 + 1024;  // layer-1 h of stack top
      float l0 = scb[0], l1 = scb[1];
      for (int k=0;k<1024;++k){ float h=htop[k]; l0 += h*scw[k]; l1 += h*scw[1024+k]; }
      fb[F_LREC+2*t]=l0; fb[F_LREC+2*t+1]=l1;
      int act = (sp<2) ? 0 : ((bi>=NSEQ) ? 1 : ((l1>l0)?1:0));
      fb[F_AREC+t] = (float)act;
      ctl[2] = (unsigned)act;
      int bcl = bi < (NSEQ-1) ? bi : (NSEQ-1);
      ctl[3] = (unsigned)idx[bcl];
    }
    gsync(cnt, gen);
    const int act = (int)((volatile unsigned*)ctl)[2];
    const int sp  = (int)((volatile unsigned*)ctl)[0];
    const int bi  = (int)((volatile unsigned*)ctl)[1];
    const int base = act ? (sp-2) : sp;       // LSTM h/c source index
    const int dst  = act ? (sp-1) : (sp+1);   // LSTM h/c destination index

    if (act == 0){
      // ---- Phase B0 (shift): embed row -> stack & vec0; prep h_prev; init gates ----
      const int row = (int)((volatile unsigned*)ctl)[3];
      const float* xrow = embed + (size_t)row*1024;
      for (int i=gtid; i<1056; i+=NT){
        float v;
        if (i<1024){ v = xrow[i]; fb[F_STK + (size_t)sp*1024 + i] = v; }
        else v = (i==1024) ? 1.0f : 0.0f;     // act_shift = [1,0], pad zeros
        bsplit(v, vb+V_X0H+i, vb+V_X0L+i);
      }
      for (int i=gtid; i<1024; i+=NT){
        bsplit(fb[F_HST+(size_t)base*2048+i],      vb+V_P0H+i, vb+V_P0L+i);
        bsplit(fb[F_HST+(size_t)base*2048+1024+i], vb+V_P1H+i, vb+V_P1L+i);
      }
      for (int i=gtid; i<4096; i+=NT){ fb[F_G0+i]=fb[F_B0g+i]; fb[F_G1+i]=fb[F_B1g+i]; }
      gsync(cnt, gen);
    } else {
      // ---- Phase B1a (reduce): assemble z, seed rpre/resv with biases, prep h_prev ----
      const int p = sp;
      for (int i=gtid; i<6144; i+=NT){
        float v;
        if      (i<2048) v = fb[F_CST + (size_t)p*2048 + i];
        else if (i<3072) v = fb[F_STK + (size_t)(p-1)*1024 + (i-2048)];
        else if (i<5120) v = fb[F_CST + (size_t)(p-1)*2048 + (i-3072)];
        else             v = fb[F_STK + (size_t)(p-2)*1024 + (i-5120)];
        bsplit(v, vb+V_ZH+i, vb+V_ZL+i);
      }
      for (int i=gtid; i<1024; i+=NT){
        fb[F_RPRE+i]=red_b1[i]; fb[F_RESV+i]=res_b[i];
        bsplit(fb[F_HST+(size_t)base*2048+i],      vb+V_P0H+i, vb+V_P0L+i);
        bsplit(fb[F_HST+(size_t)base*2048+1024+i], vb+V_P1H+i, vb+V_P1L+i);
      }
      for (int i=gtid; i<4096; i+=NT){ fb[F_G0+i]=fb[F_B0g+i]; fb[F_G1+i]=fb[F_B1g+i]; }
      gsync(cnt, gen);
      // ---- B1b: rpre += red_w1 (1024x6144) z ; 64 strips x 8 K-splits = 512 tasks ----
      for (int tk=wave; tk<512; tk+=NWAVES){
        int s = tk>>3, kg = tk&7;
        gemv_tiles(wb+O_REDH, wb+O_REDL, vb+V_ZH, vb+V_ZL, fb+F_RPRE,
                   s, kg*24, kg*24+24, 192, lane);
      }
      gsync(cnt, gen);
      // ---- B1c: split rpre for res GEMV ----
      for (int i=gtid; i<1024; i+=NT) bsplit(fb[F_RPRE+i], vb+V_RH+i, vb+V_RL+i);
      gsync(cnt, gen);
      // ---- B1d: resv += res_w (1024x1024) rpre ----
      for (int tk=wave; tk<512; tk+=NWAVES){
        int s = tk>>3, kg = tk&7;
        gemv_tiles(wb+O_RESH, wb+O_RESL, vb+V_RH, vb+V_RL, fb+F_RESV,
                   s, kg*4, kg*4+4, 32, lane);
      }
      gsync(cnt, gen);
      // ---- B1e: r = rpre + relu(resv) -> stack[p-2], vec0 = [r, 0, 1, pad] ----
      for (int i=gtid; i<1056; i+=NT){
        float v;
        if (i<1024){
          float rp = fb[F_RPRE+i], rv = fb[F_RESV+i];
          v = rp + (rv>0.0f ? rv : 0.0f);
          fb[F_STK + (size_t)(p-2)*1024 + i] = v;
        } else v = (i==1025) ? 1.0f : 0.0f;   // act_reduce = [0,1]
        bsplit(v, vb+V_X0H+i, vb+V_X0L+i);
      }
      gsync(cnt, gen);
    }

    // ---- Phase C: layer-0 gates = w_ih0*vec0 + w_hh0*h_prev0 (+biases, seeded) ----
    for (int tk=wave; tk<512; tk+=NWAVES){
      if (tk<256) gemv_tiles(wb+O_IH0H, wb+O_IH0L, vb+V_X0H, vb+V_X0L, fb+F_G0, tk,     0, 33, 33, lane);
      else        gemv_tiles(wb+O_HH0H, wb+O_HH0L, vb+V_P0H, vb+V_P0L, fb+F_G0, tk-256, 0, 32, 32, lane);
    }
    gsync(cnt, gen);

    // ---- Phase D: layer-0 activations -> h0n,c0n; split h0n for layer-1 input ----
    for (int i=gtid; i<1024; i+=NT){
      float gi=fb[F_G0+i], gf=fb[F_G0+1024+i], gg=fb[F_G0+2048+i], go=fb[F_G0+3072+i];
      float c  = fb[F_CST+(size_t)base*2048+i];
      float c2 = sigm(gf)*c + sigm(gi)*tanhf(gg);
      float h  = sigm(go)*tanhf(c2);
      fb[F_CST+(size_t)dst*2048+i] = c2;
      fb[F_HST+(size_t)dst*2048+i] = h;
      bsplit(h, vb+V_I1H+i, vb+V_I1L+i);
    }
    gsync(cnt, gen);

    // ---- Phase E: layer-1 gates = w_ih1*h0n + w_hh1*h_prev1 ----
    for (int tk=wave; tk<512; tk+=NWAVES){
      if (tk<256) gemv_tiles(wb+O_IH1H, wb+O_IH1L, vb+V_I1H, vb+V_I1L, fb+F_G1, tk,     0, 32, 32, lane);
      else        gemv_tiles(wb+O_HH1H, wb+O_HH1L, vb+V_P1H, vb+V_P1L, fb+F_G1, tk-256, 0, 32, 32, lane);
    }
    gsync(cnt, gen);

    // ---- Phase F: layer-1 activations; update sp/bi ----
    for (int i=gtid; i<1024; i+=NT){
      float gi=fb[F_G1+i], gf=fb[F_G1+1024+i], gg=fb[F_G1+2048+i], go=fb[F_G1+3072+i];
      float c  = fb[F_CST+(size_t)base*2048+1024+i];
      float c2 = sigm(gf)*c + sigm(gi)*tanhf(gg);
      float h  = sigm(go)*tanhf(c2);
      fb[F_CST+(size_t)dst*2048+1024+i] = c2;
      fb[F_HST+(size_t)dst*2048+1024+i] = h;
    }
    if (bid==0 && tid==0){
      if (act){ ctl[0] = (unsigned)(sp-1); }
      else    { ctl[0] = (unsigned)(sp+1); ctl[1] = (unsigned)(bi+1); }
    }
    gsync(cnt, gen);
  }

  // ---- epilogue: out = [stack[0] (1024) | act_rec (511) | logits (511*2)] ----
  if (bid == 0){
    for (int i=tid; i<1024;     i+=256) out[i]               = fb[F_STK+i];
    for (int i=tid; i<TSTEPS;   i+=256) out[1024+i]          = fb[F_AREC+i];
    for (int i=tid; i<2*TSTEPS; i+=256) out[1024+TSTEPS+i]   = fb[F_LREC+i];
  }
}

extern "C" void kernel_launch(void* const* d_in, const int* in_sizes, int n_in,
                              void* d_out, int out_size, void* d_ws, size_t ws_size,
                              hipStream_t stream) {
  (void)in_sizes; (void)n_in; (void)out_size; (void)ws_size;
  const int*   idx    = (const int*)  d_in[0];
  const float* embed  = (const float*)d_in[1];
  const float* w_ih0  = (const float*)d_in[2];
  const float* w_hh0  = (const float*)d_in[3];
  const float* b_ih0  = (const float*)d_in[4];
  const float* b_hh0  = (const float*)d_in[5];
  const float* w_ih1  = (const float*)d_in[6];
  const float* w_hh1  = (const float*)d_in[7];
  const float* b_ih1  = (const float*)d_in[8];
  const float* b_hh1  = (const float*)d_in[9];
  const float* h0     = (const float*)d_in[10];
  const float* c0     = (const float*)d_in[11];
  const float* scw    = (const float*)d_in[12];
  const float* scb    = (const float*)d_in[13];
  const float* red_w1 = (const float*)d_in[14];
  const float* red_b1 = (const float*)d_in[15];
  const float* res_w  = (const float*)d_in[16];
  const float* res_b  = (const float*)d_in[17];
  float* out = (float*)d_out;

  auto rp = [&](const float* W, size_t ohi, size_t olo, int cols, int chunks, int ntiles){
    int thr = ntiles*32;
    srenc_repack<<<(thr+255)/256, 256, 0, stream>>>(W, d_ws,
        (unsigned long long)ohi, (unsigned long long)olo, cols, chunks, ntiles);
  };
  rp(w_ih0,  O_IH0H, O_IH0L, 1026, 33, 256*33);
  rp(w_hh0,  O_HH0H, O_HH0L, 1024, 32, 256*32);
  rp(w_ih1,  O_IH1H, O_IH1L, 1024, 32, 256*32);
  rp(w_hh1,  O_HH1H, O_HH1L, 1024, 32, 256*32);
  rp(red_w1, O_REDH, O_REDL, 6144, 192, 64*192);
  rp(res_w,  O_RESH, O_RESL, 1024, 32,  64*32);

  srenc_init<<<2056, 256, 0, stream>>>(b_ih0, b_hh0, b_ih1, b_hh1, h0, c0, d_ws);

  srenc_main<<<NB, 256, 0, stream>>>(idx, embed, scw, scb, red_b1, res_b, d_ws, out);
}
